// MoEFFN_85126251807534
// MI455X (gfx1250) — compile-verified, matrix operand order checked by reading, not run on hardware
//
#include <hip/hip_runtime.h>
#include <hip/hip_bf16.h>
#include <math.h>

#define C_DIM 1024
#define E_DIM 8
#define H_DIM 4096
#define TOK_TILE 64

typedef __attribute__((ext_vector_type(16))) __bf16 bf16x16;
typedef __attribute__((ext_vector_type(8)))  float  f32x8;

static __device__ __forceinline__ unsigned short f2bf(float f) {
    union { float f; unsigned u; } v; v.f = f;
    unsigned r = v.u + 0x7FFFu + ((v.u >> 16) & 1u);   // round-to-nearest-even
    return (unsigned short)(r >> 16);
}

// A-operand (16x32 bf16): lane L holds row m=L%16; with half=L/16 its 16 elems are
// A[m, kbase+8h .. +7] then A[m, kbase+16+8h .. +7]  -> two aligned 16B chunks.
static __device__ __forceinline__ bf16x16 make_a(const unsigned short* row, int kbase, int half) {
    union { bf16x16 v; uint4 q[2]; } u;
    u.q[0] = *(const uint4*)(row + kbase + half * 8);
    u.q[1] = *(const uint4*)(row + kbase + 16 + half * 8);
    return u.v;
}

// B-operand: pre-swizzled in workspace, 32 contiguous bytes per lane.
static __device__ __forceinline__ bf16x16 load_b16(const unsigned short* p) {
    union { bf16x16 v; uint4 q[2]; } u;
    u.q[0] = *(const uint4*)(p);
    u.q[1] = *(const uint4*)(p + 8);
    return u.v;
}

static __device__ __forceinline__ f32x8 wmma_bf16(bf16x16 a, bf16x16 b, f32x8 c) {
    return __builtin_amdgcn_wmma_f32_16x16x32_bf16(false, a, false, b, (short)0, c, false, false);
}

// ---------------- kernel 0: zero output + control ----------------
__global__ void moe_zero(float* __restrict__ out, size_t n, int* __restrict__ ctrl) {
    size_t i = (size_t)blockIdx.x * blockDim.x + threadIdx.x;
    size_t stride = (size_t)gridDim.x * blockDim.x;
    for (size_t k = i; k < n; k += stride) out[k] = 0.0f;
    if (i < 64) ctrl[i] = 0;
}

// ---------------- kernel 1: pre-convert x to bf16 rows ----------------
__global__ void moe_pack_x(const float* __restrict__ x, unsigned short* __restrict__ xb, size_t n8) {
    size_t i = (size_t)blockIdx.x * blockDim.x + threadIdx.x;
    if (i >= n8) return;
    const float4* s = (const float4*)x + i * 2;
    float4 a = s[0], b = s[1];
    union { unsigned short h[8]; uint4 q; } u;
    u.h[0] = f2bf(a.x); u.h[1] = f2bf(a.y); u.h[2] = f2bf(a.z); u.h[3] = f2bf(a.w);
    u.h[4] = f2bf(b.x); u.h[5] = f2bf(b.y); u.h[6] = f2bf(b.z); u.h[7] = f2bf(b.w);
    *(uint4*)(xb + i * 8) = u.q;
}

// ---------------- kernels 2/3: pack weights to bf16 in WMMA-B lane layout ----
// w1t layout: slot s = ((e*(H/16) + ntg)*(C/32) + kt)*32 + lane, 16 bf16 per slot.
__global__ void moe_pack_w1(const float* __restrict__ w1, unsigned short* __restrict__ w1t) {
    int s = blockIdx.x * blockDim.x + threadIdx.x;
    if (s >= 8 * 256 * 32 * 32) return;
    int lane = s & 31;
    int kt   = (s >> 5)  & 31;    // C/32
    int ntg  = (s >> 10) & 255;   // H/16
    int e    =  s >> 18;
    int half = lane >> 4;
    int n    = ntg * 16 + (lane & 15);
    int kb   = kt * 32 + half * 8;
    unsigned short* dst = w1t + (size_t)s * 16;
#pragma unroll
    for (int j = 0; j < 8; ++j) {
        dst[j]     = f2bf(w1[((size_t)e * C_DIM + kb + j)      * H_DIM + n]);
        dst[8 + j] = f2bf(w1[((size_t)e * C_DIM + kb + 16 + j) * H_DIM + n]);
    }
}

// w2t layout: slot s = ((e*(C/16) + ntg)*(H/32) + kt)*32 + lane
__global__ void moe_pack_w2(const float* __restrict__ w2, unsigned short* __restrict__ w2t) {
    int s = blockIdx.x * blockDim.x + threadIdx.x;
    if (s >= 8 * 64 * 128 * 32) return;
    int lane = s & 31;
    int kt   = (s >> 5)  & 127;   // H/32
    int ntg  = (s >> 12) & 63;    // C/16
    int e    =  s >> 18;
    int half = lane >> 4;
    int n    = ntg * 16 + (lane & 15);
    int kb   = kt * 32 + half * 8;
    unsigned short* dst = w2t + (size_t)s * 16;
#pragma unroll
    for (int j = 0; j < 8; ++j) {
        dst[j]     = f2bf(w2[((size_t)e * H_DIM + kb + j)      * C_DIM + n]);
        dst[8 + j] = f2bf(w2[((size_t)e * H_DIM + kb + 16 + j) * C_DIM + n]);
    }
}

// ---------------- kernel 4: fp32 router (logits, top-2, softmax, counts) -----
__global__ void moe_router(const float* __restrict__ x, const float* __restrict__ gw, int N,
                           int* __restrict__ topIdx, float* __restrict__ topGate,
                           int* __restrict__ ctrl) {
    int n = blockIdx.x * blockDim.x + threadIdx.x;
    if (n >= N) return;
    float acc[E_DIM];
#pragma unroll
    for (int e = 0; e < E_DIM; ++e) acc[e] = 0.0f;
    const float* xr = x + (size_t)n * C_DIM;
    for (int c = 0; c < C_DIM; ++c) {
        float xv = xr[c];
        const float4* g4 = (const float4*)(gw + (size_t)c * E_DIM);
        float4 a = g4[0], b = g4[1];
        acc[0] = fmaf(xv, a.x, acc[0]); acc[1] = fmaf(xv, a.y, acc[1]);
        acc[2] = fmaf(xv, a.z, acc[2]); acc[3] = fmaf(xv, a.w, acc[3]);
        acc[4] = fmaf(xv, b.x, acc[4]); acc[5] = fmaf(xv, b.y, acc[5]);
        acc[6] = fmaf(xv, b.z, acc[6]); acc[7] = fmaf(xv, b.w, acc[7]);
    }
    int i0 = 0;
#pragma unroll
    for (int e = 1; e < E_DIM; ++e) if (acc[e] > acc[i0]) i0 = e;
    int i1 = (i0 == 0) ? 1 : 0;
#pragma unroll
    for (int e = 0; e < E_DIM; ++e)
        if (e != i0 && acc[e] > acc[i1]) i1 = e;
    float v0 = acc[i0], v1 = acc[i1];
    float e1 = expf(v1 - v0);                 // e0 = 1
    float inv = 1.0f / (1.0f + e1);
    topIdx[2 * n]     = i0;  topGate[2 * n]     = inv;
    topIdx[2 * n + 1] = i1;  topGate[2 * n + 1] = e1 * inv;
    atomicAdd(&ctrl[i0], 1);
    atomicAdd(&ctrl[i1], 1);
}

// ---------------- kernel 5: scan (offsets + tile offsets) --------------------
// ctrl: [0..7]=counts, [8..15]=cursors, [16..24]=offsets, [32..40]=tileOffsets
__global__ void moe_scan(int* __restrict__ ctrl) {
    if (threadIdx.x == 0 && blockIdx.x == 0) {
        int off = 0, toff = 0;
        ctrl[16] = 0; ctrl[32] = 0;
        for (int e = 0; e < E_DIM; ++e) {
            off  += ctrl[e];
            toff += (ctrl[e] + TOK_TILE - 1) / TOK_TILE;
            ctrl[16 + e + 1] = off;
            ctrl[32 + e + 1] = toff;
        }
    }
}

// ---------------- kernel 6: assignment (token -> expert slots) ---------------
__global__ void moe_assign(const int* __restrict__ topIdx, const float* __restrict__ topGate,
                           int N, int* __restrict__ ctrl,
                           int* __restrict__ permTok, float* __restrict__ permGate) {
    int n = blockIdx.x * blockDim.x + threadIdx.x;
    if (n >= N) return;
#pragma unroll
    for (int k = 0; k < 2; ++k) {
        int e = topIdx[2 * n + k];
        int pos = atomicAdd(&ctrl[8 + e], 1);
        int slot = ctrl[16 + e] + pos;
        permTok[slot]  = n;
        permGate[slot] = topGate[2 * n + k];
    }
}

// ---------------- kernel 7: fused grouped FFN (WMMA bf16, 64-token tiles) ----
__global__ __launch_bounds__(256, 1)
void moe_ffn(const unsigned short* __restrict__ xb,
             const float* __restrict__ b1, const float* __restrict__ b2,
             const unsigned short* __restrict__ w1t, const unsigned short* __restrict__ w2t,
             const int* __restrict__ permTok, const float* __restrict__ permGate,
             const int* __restrict__ ctrl, float* __restrict__ out) {
    __shared__ __align__(16) unsigned short xs[TOK_TILE][C_DIM + 8]; // bf16, padded stride
    __shared__ __align__(16) unsigned short hs[TOK_TILE][64 + 8];    // gelu(h) chunk
    __shared__ int   token_l[TOK_TILE];
    __shared__ float gate_l[TOK_TILE];

    const int b = blockIdx.x;
    const int* tileOff = ctrl + 32;
    if (b >= tileOff[8]) return;

    int e = 0;
    while (e < E_DIM - 1 && b >= tileOff[e + 1]) ++e;
    const int tIdx = b - tileOff[e];
    const int cnt  = ctrl[e];
    const int off  = ctrl[16 + e];
    const int rows = min(TOK_TILE, cnt - tIdx * TOK_TILE);

    const int tid  = threadIdx.x;
    const int lane = tid & 31;
    const int w    = tid >> 5;       // 8 waves
    const int half = lane >> 4;
    const int ln   = lane & 15;

    if (tid < TOK_TILE) {
        if (tid < rows) {
            token_l[tid] = permTok[off + tIdx * TOK_TILE + tid];
            gate_l[tid]  = permGate[off + tIdx * TOK_TILE + tid];
        } else {
            token_l[tid] = 0;
            gate_l[tid]  = 0.0f;
        }
    }
    __syncthreads();

    // ---- stage x rows into LDS via CDNA5 async copy (ASYNCcnt-tracked) ----
    {
        const unsigned xs_base = (unsigned)(uintptr_t)(&xs[0][0]);
#pragma unroll 4
        for (int i = 0; i < 32; ++i) {
            int chunk = tid + 256 * i;        // 8192 chunks of 16B (64 rows x 128)
            int row   = chunk >> 7;
            int cseg  = chunk & 127;
            if (row < rows) {
                const unsigned short* src = xb + (size_t)token_l[row] * C_DIM + cseg * 8;
                unsigned dst = xs_base + (unsigned)(row * ((C_DIM + 8) * 2) + cseg * 16);
                asm volatile("global_load_async_to_lds_b128 %0, %1, off"
                             :: "v"(dst), "v"(src) : "memory");
            } else {
                uint4 z; z.x = z.y = z.z = z.w = 0u;
                *(uint4*)(&xs[0][0] + (size_t)row * (C_DIM + 8) + cseg * 8) = z;
            }
        }
        asm volatile("s_wait_asynccnt 0x0" ::: "memory");
    }
    __syncthreads();

    // wave w: phase-1 tiles (mt in {mtp, mtp+1}, nt1);  phase-2 owns cols [w*128, w*128+128)
    const int nt1 = w >> 1;            // 0..3
    const int mtp = (w & 1) * 2;       // 0 or 2

    f32x8 oacc[4][8];
#pragma unroll
    for (int mt = 0; mt < 4; ++mt)
#pragma unroll
        for (int nt = 0; nt < 8; ++nt)
#pragma unroll
            for (int r = 0; r < 8; ++r) oacc[mt][nt][r] = 0.0f;

    const unsigned short* xrow0 = &xs[mtp * 16 + ln][0];
    const unsigned short* xrow1 = &xs[(mtp + 1) * 16 + ln][0];

    for (int hc = 0; hc < H_DIM / 64; ++hc) {
        // ---- phase 1: h = x @ w1 chunk, K = C = 1024; one B feeds two WMMAs ----
        f32x8 hacc0, hacc1;
#pragma unroll
        for (int r = 0; r < 8; ++r) { hacc0[r] = 0.0f; hacc1[r] = 0.0f; }
        const int ntg1 = hc * 4 + nt1;
        const size_t w1base = ((size_t)(e * 256 + ntg1) * 32) * 512 + (size_t)lane * 16;
        for (int kt = 0; kt < 32; ++kt) {
            bf16x16 bm = load_b16(w1t + w1base + (size_t)kt * 512);
            bf16x16 a0 = make_a(xrow0, kt * 32, half);
            bf16x16 a1 = make_a(xrow1, kt * 32, half);
            hacc0 = wmma_bf16(a0, bm, hacc0);
            hacc1 = wmma_bf16(a1, bm, hacc1);
        }
        // bias + exact GELU, write bf16 chunk to LDS
        const int hcolg = hc * 64 + nt1 * 16 + ln;
        const float bias1 = b1[(size_t)e * H_DIM + hcolg];
#pragma unroll
        for (int r = 0; r < 8; ++r) {
            float v0 = hacc0[r] + bias1;
            float v1 = hacc1[r] + bias1;
            float g0 = 0.5f * v0 * (1.0f + erff(v0 * 0.70710678118654752f));
            float g1 = 0.5f * v1 * (1.0f + erff(v1 * 0.70710678118654752f));
            hs[mtp * 16 + r + half * 8][nt1 * 16 + ln]       = f2bf(g0);
            hs[(mtp + 1) * 16 + r + half * 8][nt1 * 16 + ln] = f2bf(g1);
        }
        __syncthreads();

        // ---- phase 2: out += h_chunk @ w2 chunk, K = 64; one B feeds 4 WMMAs ----
#pragma unroll
        for (int kt2 = 0; kt2 < 2; ++kt2) {
            bf16x16 A[4];
#pragma unroll
            for (int mt = 0; mt < 4; ++mt)
                A[mt] = make_a(&hs[mt * 16 + ln][0], kt2 * 32, half);
            const int ktH = hc * 2 + kt2;
#pragma unroll
            for (int nt = 0; nt < 8; ++nt) {
                const int ntg2 = w * 8 + nt;
                const size_t w2off = ((size_t)(e * 64 + ntg2) * 128 + ktH) * 512 + (size_t)lane * 16;
                bf16x16 bm = load_b16(w2t + w2off);
#pragma unroll
                for (int mt = 0; mt < 4; ++mt)
                    oacc[mt][nt] = wmma_bf16(A[mt], bm, oacc[mt][nt]);
            }
        }
        __syncthreads();
    }

    // ---- epilogue: out[token] += gate * (acc + b2) ----
#pragma unroll
    for (int mt = 0; mt < 4; ++mt) {
#pragma unroll
        for (int nt = 0; nt < 8; ++nt) {
            const int col = w * 128 + nt * 16 + ln;
            const float bias2 = b2[(size_t)e * C_DIM + col];
#pragma unroll
            for (int r = 0; r < 8; ++r) {
                int row = mt * 16 + r + half * 8;
                if (row < rows) {
                    float val = (oacc[mt][nt][r] + bias2) * gate_l[row];
                    atomicAdd(out + (size_t)token_l[row] * C_DIM + col, val);
                }
            }
        }
    }
}

// ---------------- host launch ----------------
extern "C" void kernel_launch(void* const* d_in, const int* in_sizes, int n_in,
                              void* d_out, int out_size, void* d_ws, size_t ws_size,
                              hipStream_t stream) {
    const float* x    = (const float*)d_in[0];
    const float* gw   = (const float*)d_in[1];
    const float* w1   = (const float*)d_in[2];
    const float* b1   = (const float*)d_in[3];
    const float* w2   = (const float*)d_in[4];
    const float* b2   = (const float*)d_in[5];
    float* out = (float*)d_out;

    const int N = in_sizes[0] / C_DIM;          // 8192 tokens

    const size_t W1T_ELEMS = (size_t)8 * 256 * 32 * 32 * 16;  // 33,554,432 bf16
    const size_t W2T_ELEMS = (size_t)8 * 64 * 128 * 32 * 16;
    unsigned short* w1t = (unsigned short*)d_ws;
    unsigned short* w2t = w1t + W1T_ELEMS;
    unsigned short* xb  = w2t + W2T_ELEMS;                     // N*C bf16 (16 MB)
    char* p = (char*)(xb + (size_t)N * C_DIM);
    int*   permTok  = (int*)p;    p += (size_t)2 * N * sizeof(int);
    float* permGate = (float*)p;  p += (size_t)2 * N * sizeof(float);
    int*   topIdx   = (int*)p;    p += (size_t)2 * N * sizeof(int);
    float* topGate  = (float*)p;  p += (size_t)2 * N * sizeof(float);
    int*   ctrl     = (int*)p;    // 64 ints

    moe_zero<<<2048, 256, 0, stream>>>(out, (size_t)N * C_DIM, ctrl);
    moe_pack_x<<<(int)(((size_t)N * C_DIM / 8 + 255) / 256), 256, 0, stream>>>(
        x, xb, (size_t)N * C_DIM / 8);
    moe_pack_w1<<<(8 * 256 * 32 * 32) / 256, 256, 0, stream>>>(w1, w1t);
    moe_pack_w2<<<(8 * 64 * 128 * 32) / 256, 256, 0, stream>>>(w2, w2t);
    moe_router<<<(N + 255) / 256, 256, 0, stream>>>(x, gw, N, topIdx, topGate, ctrl);
    moe_scan<<<1, 1, 0, stream>>>(ctrl);
    moe_assign<<<(N + 255) / 256, 256, 0, stream>>>(topIdx, topGate, N, ctrl, permTok, permGate);

    const int maxTiles = (2 * N) / TOK_TILE + E_DIM;  // 264 upper bound
    moe_ffn<<<maxTiles, 256, 0, stream>>>(xb, b1, b2, w1t, w2t, permTok, permGate, ctrl, out);
}
